// SoftTripletGraph_50895362457699
// MI455X (gfx1250) — matrix operand order repaired and checked by reading, not compiled, not run on hardware
//
#include <hip/hip_runtime.h>
#include <hip/hip_bf16.h>

// ---------------------------------------------------------------------------
// SoftTripletGraph on MI455X (gfx1250, wave32, WMMA bf16 16x16x32)
// B=32, L=4096, H=256, N=512, M = B*N = 16384
// All WMMA operand fragments are loaded as 16-byte vector loads: A-side
// directly (k order 16g+e gives one contiguous 16-element run per lane), B-side via
// fragment-major pre-swizzled buffers (weights: tiny swizzle kernel; node:
// LDS-staged transpose-swizzle).
// ---------------------------------------------------------------------------

#define Bc 32
#define Lc 4096
#define Hc 256
#define Nc 512
#define Mc (Bc * Nc)   // 16384

typedef __attribute__((ext_vector_type(16))) __bf16 v16bf;
typedef __attribute__((ext_vector_type(8)))  __bf16 v8bf;
typedef __attribute__((ext_vector_type(8)))  float  v8f;

__device__ __forceinline__ v16bf join16(v8bf lo, v8bf hi) {
  return __builtin_shufflevector(lo, hi, 0, 1, 2, 3, 4, 5, 6, 7, 8, 9, 10, 11,
                                 12, 13, 14, 15);
}

// A (16x32 bf16, MxK) from row-major src. The lane's 16 elements are one
// contiguous K-run: [16g .. 16g+15] (matches the B-side k order).
__device__ __forceinline__ v16bf frag_a(const __bf16* __restrict__ p, int ld,
                                        int row0, int k0, int lane) {
  const int g = lane >> 4, m = lane & 15;
  const __bf16* q = p + (size_t)(row0 + m) * ld + k0 + (g << 4);
  return join16(*(const v8bf*)q, *(const v8bf*)(q + 8));
}

// B = P^T where P is row-major: element (k=16g+e, col) = P[col, k] -> the
// lane's 16 elements are contiguous in memory.
__device__ __forceinline__ v16bf frag_bt(const __bf16* __restrict__ p, int ld,
                                         int k0, int col0, int lane) {
  const __bf16* q = p + (size_t)(col0 + (lane & 15)) * ld + k0 +
                    ((lane >> 4) << 4);
  return join16(*(const v8bf*)q, *(const v8bf*)(q + 8));
}

// B fragment from a fragment-major swizzled buffer:
//   swz[((kt*ntc + ct)*32 + lane)*16 + e]
__device__ __forceinline__ v16bf frag_bs(const __bf16* __restrict__ p, int ntc,
                                         int kt, int ct, int lane) {
  const __bf16* q = p + (((size_t)(kt * ntc + ct) * 32 + lane) << 4);
  return join16(*(const v8bf*)q, *(const v8bf*)(q + 8));
}

__device__ __forceinline__ v8f wmma_bf16(v16bf a, v16bf b, v8f c) {
  return __builtin_amdgcn_wmma_f32_16x16x32_bf16(false, a, false, b,
                                                 (short)0, c, false, false);
}

// ------------------------- 0) bulk copy out = embeddings -------------------
__global__ void k_copy(const float4* __restrict__ src, float4* __restrict__ dst,
                       int n) {
  const int stride = gridDim.x * blockDim.x;
  for (int i = blockIdx.x * blockDim.x + threadIdx.x; i < n; i += stride) {
    __builtin_prefetch(src + i + stride, 0, 0);   // global_prefetch_b8
    dst[i] = src[i];
  }
}

__global__ void k_init(int* b_has_edge) {
  if (threadIdx.x < Bc) b_has_edge[threadIdx.x] = 0;
}

// --------- 1a) weight f32 -> fragment-major bf16 swizzle (B operand) -------
// W: [K x C] row-major f32. One block per 32x16 tile; lane writes its 16
// contiguous fragment elements (2 x b128 stores).
__global__ __launch_bounds__(32) void k_swz_weight(const float* __restrict__ W,
                                                   int C, __bf16* __restrict__ swz) {
  const int lane = threadIdx.x;
  const int ntc = C >> 4;
  const int kt = blockIdx.x / ntc, ct = blockIdx.x - kt * ntc;
  const int g = lane >> 4, n = lane & 15;
  __bf16 tmp[16];
#pragma unroll
  for (int e = 0; e < 16; ++e)
    tmp[e] = (__bf16)W[(size_t)(kt * 32 + g * 16 + e) * C + ct * 16 + n];
  __bf16* q = swz + (((size_t)blockIdx.x * 32 + lane) << 4);
#pragma unroll
  for (int e = 0; e < 16; ++e) q[e] = tmp[e];
}

// --------- 1b) edge precomputation -----------------------------------------
// ae[t] = sum_h leaky(edge[t,h]) * W_attn[2H+h]   (b_attn cancels in softmax)
// ee_g[t,h] = sum_k edge[t,k] * W_gat[256+k, h]
__global__ __launch_bounds__(256) void k_edge_prep(
    const float* __restrict__ edge_emb, const float* __restrict__ W_attn,
    const float* __restrict__ W_gat, float* __restrict__ ae,
    float* __restrict__ ee_g) {
  const int h = threadIdx.x;
  float s0 = 0.f, s1 = 0.f;
  for (int k = 0; k < Hc; ++k) {
    const float w = W_gat[(size_t)(Hc + k) * Hc + h];
    s0 += edge_emb[k] * w;
    s1 += edge_emb[Hc + k] * w;
  }
  ee_g[h] = s0;
  ee_g[Hc + h] = s1;
  if (h < 2) {
    float s = 0.f;
    for (int k = 0; k < Hc; ++k) {
      const float x = edge_emb[h * Hc + k];
      s += (x >= 0.f ? x : 0.2f * x) * W_attn[2 * Hc + k];
    }
    ae[h] = s;
  }
}

// ------------------------- 2) span means + packed span keys ----------------
// Spans only touch the first ~135 rows per batch -> L2 resident.
__global__ __launch_bounds__(256) void k_spans(
    const float* __restrict__ emb, const int* __restrict__ asp_st,
    const int* __restrict__ asp_len, const int* __restrict__ opi_st,
    const int* __restrict__ opi_len, __bf16* __restrict__ span,
    int* __restrict__ key) {
  const int m = blockIdx.x;          // flattened (b,n)
  const int b = m >> 9;
  const int h = threadIdx.x;
  const float* eb = emb + (size_t)b * Lc * Hc + h;
  const int as = asp_st[m], al = asp_len[m];
  const int os = opi_st[m], ol = opi_len[m];
  float sa = 0.f;
  for (int k = 0; k <= al; ++k) sa += eb[(size_t)(as + k) * Hc];
  float so = 0.f;
  for (int k = 0; k <= ol; ++k) so += eb[(size_t)(os + k) * Hc];
  span[(size_t)m * 512 + h]      = (__bf16)(sa / (float)(al + 1));
  span[(size_t)m * 512 + Hc + h] = (__bf16)(so / (float)(ol + 1));
  if (h == 0) key[m] = (as | (al << 8)) | ((os | (ol << 8)) << 16);
}

// ------------------------- 3) node = span @ W_trip + bias + onehot row -----
__global__ __launch_bounds__(32) void k_trip_gemm(
    const __bf16* __restrict__ span, const __bf16* __restrict__ wt_swz,
    const float* __restrict__ wtrip_tail, const float* __restrict__ b_trip,
    const int* __restrict__ sent_id, float* __restrict__ node_f32,
    __bf16* __restrict__ node_bf16) {
  const int lane = threadIdx.x;
  const int row0 = blockIdx.x * 16, ct = blockIdx.y;
  const int col0 = ct * 16;
  v8f acc = {};
  for (int kt = 0; kt < 512 / 32; ++kt) {
    v16bf a = frag_a(span, 512, row0, kt * 32, lane);
    v16bf b = frag_bs(wt_swz, Hc / 16, kt, ct, lane);
    acc = wmma_bf16(a, b, acc);
  }
  const int g = lane >> 4, n = lane & 15;
  const int col = col0 + n;
#pragma unroll
  for (int r = 0; r < 8; ++r) {
    const int m = row0 + g * 8 + r;
    const float v = acc[r] + b_trip[col] + wtrip_tail[sent_id[m] * Hc + col];
    node_f32[(size_t)m * Hc + col]  = v;
    node_bf16[(size_t)m * Hc + col] = (__bf16)v;
  }
}

// --------- 3b) node -> fragment-major swizzle (B operand for agg GEMM) -----
// One block per (b, kt): stage a 32(row) x 256(col) bf16 tile in LDS with
// coalesced b128 loads, then each wave emits fragment-major 32B runs.
__global__ __launch_bounds__(256) void k_swz_node(
    const __bf16* __restrict__ node, __bf16* __restrict__ swz) {
  __shared__ __bf16 tile[32 * Hc];       // 16 KB
  const int b = blockIdx.x >> 4, kt = blockIdx.x & 15;
  const __bf16* src = node + ((size_t)b * Nc + kt * 32) * Hc;
  const uint4* s4 = (const uint4*)src;
  uint4* t4 = (uint4*)tile;
  for (int i = threadIdx.x; i < 32 * Hc / 8; i += 256) t4[i] = s4[i];
  __syncthreads();
  const int wave = threadIdx.x >> 5, lane = threadIdx.x & 31;
  const int g = lane >> 4, n = lane & 15;
  for (int ct = wave; ct < Hc / 16; ct += 8) {
    __bf16 v[16];
#pragma unroll
    for (int e = 0; e < 16; ++e) v[e] = tile[(g * 16 + e) * Hc + ct * 16 + n];
    __bf16* q = swz + ((((size_t)(b * 16 + kt) * (Hc / 16) + ct) * 32 + lane) << 4);
#pragma unroll
    for (int e = 0; e < 16; ++e) q[e] = v[e];
  }
}

// ------------------------- 4) a_src[m] = leaky(node[m]) . W_attn[H:2H] -----
__global__ __launch_bounds__(256) void k_attn_proj(
    const float* __restrict__ node_f32, const float* __restrict__ W_attn,
    float* __restrict__ a_src) {
  const int wave = threadIdx.x >> 5, lane = threadIdx.x & 31;
  const int m = blockIdx.x * 8 + wave;
  const float* nr = node_f32 + (size_t)m * Hc;
  float s = 0.f;
  for (int h = lane; h < Hc; h += 32) {
    const float x = nr[h];
    s += (x >= 0.f ? x : 0.2f * x) * W_attn[Hc + h];
  }
#pragma unroll
  for (int o = 16; o; o >>= 1) s += __shfl_xor(s, o, 32);
  if (lane == 0) a_src[m] = s;
}

// ------------------------- 5) dots sign + span masks -> edge bytes ---------
// sim > 0 with THR=0 <=> dot > 0 (cosine denominator positive) -> the whole
// cosine normalization is dropped; only the Gram matrix sign is needed.
__global__ __launch_bounds__(32) void k_sim_gemm(
    const __bf16* __restrict__ node_bf16, const int* __restrict__ key,
    unsigned char* __restrict__ em) {
  const int lane = threadIdx.x;
  const int b = blockIdx.z;
  const int i0 = blockIdx.x * 16, j0 = blockIdx.y * 16;
  const __bf16* nb = node_bf16 + (size_t)b * Nc * Hc;
  v8f acc = {};
  for (int k = 0; k < Hc; k += 32) {
    v16bf a  = frag_a(nb, Hc, i0, k, lane);
    v16bf bm = frag_bt(nb, Hc, k, j0, lane);
    acc = wmma_bf16(a, bm, acc);
  }
  const int g = lane >> 4, n = lane & 15;
  const int j = j0 + n;
  const int kj = key[b * Nc + j];
#pragma unroll
  for (int r = 0; r < 8; ++r) {
    const int i = i0 + g * 8 + r;
    const int ki = key[b * Nc + i];
    unsigned char mbit = 0;
    if (acc[r] > 0.f && i != j) {
      if ((ki & 0xFFFF) == (kj & 0xFFFF)) mbit |= 1;   // same aspect span
      if ((ki >> 16) == (kj >> 16))       mbit |= 2;   // same opinion span
    }
    em[((size_t)b * Nc + i) * Nc + j] = mbit;
  }
}

// ------------------------- 6) masked softmax over (src, type) --------------
// a_tgt[i] and b_attn are row-constants -> cancel under softmax.
__global__ __launch_bounds__(256) void k_softmax(
    const unsigned char* __restrict__ em, const float* __restrict__ a_src,
    const float* __restrict__ ae, __bf16* __restrict__ w_bf16,
    float* __restrict__ wsum_e, unsigned char* __restrict__ has_neigh,
    int* __restrict__ b_has_edge) {
  const int wave = threadIdx.x >> 5, lane = threadIdx.x & 31;
  const int row = blockIdx.x * 8 + wave;   // flattened (b,i)
  const int b = row >> 9;
  const unsigned char* er = em + (size_t)row * Nc;
  const float* as = a_src + (size_t)b * Nc;
  const float ae0 = ae[0], ae1 = ae[1];
  float mx = -3.0e38f;
  int any = 0;
  for (int j = lane; j < Nc; j += 32) {
    const int mb = er[j];
    if (mb) {
      any = 1;
      const float s = as[j];
      if (mb & 1) mx = fmaxf(mx, s + ae0);
      if (mb & 2) mx = fmaxf(mx, s + ae1);
    }
  }
#pragma unroll
  for (int o = 16; o; o >>= 1) {
    mx = fmaxf(mx, __shfl_xor(mx, o, 32));
    any |= __shfl_xor(any, o, 32);
  }
  float s0 = 0.f, s1 = 0.f;
  __bf16* wr = w_bf16 + (size_t)row * Nc;
  if (any) {
    float z = 0.f;
    for (int j = lane; j < Nc; j += 32) {
      const int mb = er[j];
      if (mb) {
        const float s = as[j];
        if (mb & 1) z += __expf(s + ae0 - mx);
        if (mb & 2) z += __expf(s + ae1 - mx);
      }
    }
#pragma unroll
    for (int o = 16; o; o >>= 1) z += __shfl_xor(z, o, 32);
    const float rz = 1.f / z;
    for (int j = lane; j < Nc; j += 32) {
      const int mb = er[j];
      float w0 = 0.f, w1 = 0.f;
      if (mb) {
        const float s = as[j];
        if (mb & 1) w0 = __expf(s + ae0 - mx) * rz;
        if (mb & 2) w1 = __expf(s + ae1 - mx) * rz;
      }
      wr[j] = (__bf16)(w0 + w1);
      s0 += w0;
      s1 += w1;
    }
#pragma unroll
    for (int o = 16; o; o >>= 1) {
      s0 += __shfl_xor(s0, o, 32);
      s1 += __shfl_xor(s1, o, 32);
    }
  } else {
    for (int j = lane; j < Nc; j += 32) wr[j] = (__bf16)0.f;
  }
  if (lane == 0) {
    wsum_e[row * 2 + 0] = s0;
    wsum_e[row * 2 + 1] = s1;
    has_neigh[row] = (unsigned char)any;
    if (any) atomicOr(&b_has_edge[b], 1);
  }
}

// ------------------------- 7) agg_node = w @ node --------------------------
__global__ __launch_bounds__(32) void k_agg_gemm(
    const __bf16* __restrict__ w_bf16, const __bf16* __restrict__ node_swz,
    __bf16* __restrict__ agg_bf16) {
  const int lane = threadIdx.x;
  const int b = blockIdx.z;
  const int i0 = blockIdx.x * 16, ct = blockIdx.y;
  const __bf16* wa = w_bf16 + (size_t)b * Nc * Nc;
  const __bf16* ns = node_swz + (size_t)b * 16 * (Hc / 16) * 32 * 16;
  v8f acc = {};
  for (int kt = 0; kt < Nc / 32; ++kt) {
    v16bf a  = frag_a(wa, Nc, i0, kt * 32, lane);
    v16bf bm = frag_bs(ns, Hc / 16, kt, ct, lane);
    acc = wmma_bf16(a, bm, acc);
  }
  const int g = lane >> 4, n = lane & 15;
#pragma unroll
  for (int r = 0; r < 8; ++r) {
    const size_t m = (size_t)b * Nc + i0 + g * 8 + r;
    agg_bf16[m * Hc + ct * 16 + n] = (__bf16)acc[r];
  }
}

// ------- 8) upd = relu(agg@Wg1 + wsum_e@ee_g + b_gat), fused scatter -------
__global__ __launch_bounds__(32) void k_gat_gemm(
    const __bf16* __restrict__ agg_bf16, const __bf16* __restrict__ wg1_swz,
    const float* __restrict__ wsum_e, const float* __restrict__ ee_g,
    const float* __restrict__ b_gat, const unsigned char* __restrict__ has_neigh,
    const int* __restrict__ b_has_edge, const float* __restrict__ node_f32,
    const int* __restrict__ asp_st, const int* __restrict__ opi_st,
    float* __restrict__ out) {
  const int lane = threadIdx.x;
  const int row0 = blockIdx.x * 16, ct = blockIdx.y;
  v8f acc = {};
  for (int kt = 0; kt < Hc / 32; ++kt) {
    v16bf a  = frag_a(agg_bf16, Hc, row0, kt * 32, lane);
    v16bf bm = frag_bs(wg1_swz, Hc / 16, kt, ct, lane);
    acc = wmma_bf16(a, bm, acc);
  }
  const int g = lane >> 4, n = lane & 15;
  const int col = ct * 16 + n;
  const float bg = b_gat[col];
  const float e0 = ee_g[col], e1 = ee_g[Hc + col];
#pragma unroll
  for (int r = 0; r < 8; ++r) {
    const int m = row0 + g * 8 + r;
    const int b = m >> 9;
    if (!b_has_edge[b]) continue;   // batch with no edges adds nothing
    float val;
    if (has_neigh[m]) {
      const float u = acc[r] + wsum_e[m * 2] * e0 + wsum_e[m * 2 + 1] * e1 + bg;
      val = u > 0.f ? u : 0.f;
    } else {
      val = node_f32[(size_t)m * Hc + col];
    }
    const int center = (asp_st[m] + opi_st[m]) >> 1;
    atomicAdd(&out[((size_t)b * Lc + center) * Hc + col], val);
  }
}

// ---------------------------------------------------------------------------
extern "C" void kernel_launch(void* const* d_in, const int* in_sizes, int n_in,
                              void* d_out, int out_size, void* d_ws,
                              size_t ws_size, hipStream_t stream) {
  const float* emb     = (const float*)d_in[0];
  const float* W_trip  = (const float*)d_in[1];
  const float* b_trip  = (const float*)d_in[2];
  const float* edge_emb= (const float*)d_in[3];
  const float* W_attn  = (const float*)d_in[4];
  const float* W_gat   = (const float*)d_in[6];
  const float* b_gat   = (const float*)d_in[7];
  const int* asp_st    = (const int*)d_in[8];
  const int* asp_len   = (const int*)d_in[9];
  const int* opi_st    = (const int*)d_in[10];
  const int* opi_len   = (const int*)d_in[11];
  const int* sent_id   = (const int*)d_in[12];
  float* out = (float*)d_out;

  char* p = (char*)d_ws;
  auto alloc = [&](size_t bytes) -> char* {
    char* r = p;
    p += (bytes + 255) & ~(size_t)255;
    return r;
  };
  __bf16* span      = (__bf16*)alloc((size_t)Mc * 512 * 2);
  float*  node_f32  = (float*)alloc((size_t)Mc * Hc * 4);
  __bf16* node_bf16 = (__bf16*)alloc((size_t)Mc * Hc * 2);
  __bf16* node_swz  = (__bf16*)alloc((size_t)Mc * Hc * 2);
  float*  a_src     = (float*)alloc((size_t)Mc * 4);
  int*    key       = (int*)alloc((size_t)Mc * 4);
  unsigned char* em = (unsigned char*)alloc((size_t)Bc * Nc * Nc);
  __bf16* w_bf16    = (__bf16*)alloc((size_t)Bc * Nc * Nc * 2);
  float*  wsum_e    = (float*)alloc((size_t)Mc * 2 * 4);
  unsigned char* hn = (unsigned char*)alloc((size_t)Mc);
  int*    b_has_edge= (int*)alloc((size_t)Bc * 4);
  __bf16* agg_bf16  = (__bf16*)alloc((size_t)Mc * Hc * 2);
  __bf16* wt_swz    = (__bf16*)alloc((size_t)512 * Hc * 2);
  __bf16* wg1_swz   = (__bf16*)alloc((size_t)Hc * Hc * 2);
  float*  ae        = (float*)alloc(2 * 4);
  float*  ee_g      = (float*)alloc((size_t)2 * Hc * 4);

  // 0) out = embeddings (bandwidth-dominant step)
  const int n4 = Bc * Lc * Hc / 4;
  k_copy<<<8192, 256, 0, stream>>>((const float4*)emb, (float4*)out, n4);
  k_init<<<1, 32, 0, stream>>>(b_has_edge);

  // 1) weight swizzles + edge precomputation
  k_swz_weight<<<(512 / 32) * (Hc / 16), 32, 0, stream>>>(W_trip, Hc, wt_swz);
  k_swz_weight<<<(Hc / 32) * (Hc / 16), 32, 0, stream>>>(W_gat, Hc, wg1_swz);
  k_edge_prep<<<1, 256, 0, stream>>>(edge_emb, W_attn, W_gat, ae, ee_g);

  // 2) span means
  k_spans<<<Mc, 256, 0, stream>>>(emb, asp_st, asp_len, opi_st, opi_len, span,
                                  key);

  // 3) node = span @ W_trip + b_trip + onehot row   [16384 x 512 x 256]
  k_trip_gemm<<<dim3(Mc / 16, Hc / 16), 32, 0, stream>>>(
      span, wt_swz, W_trip + (size_t)512 * Hc, b_trip, sent_id, node_f32,
      node_bf16);

  // 3b) node -> fragment-major layout for use as B operand
  k_swz_node<<<Bc * (Nc / 32), 256, 0, stream>>>(node_bf16, node_swz);

  // 4) a_src (a_tgt and b_attn cancel under softmax)
  k_attn_proj<<<Mc / 8, 256, 0, stream>>>(node_f32, W_attn, a_src);

  // 5) sign(node . node^T) + span-key masks -> edge bytes  [32 x 512x512x256]
  k_sim_gemm<<<dim3(Nc / 16, Nc / 16, Bc), 32, 0, stream>>>(node_bf16, key, em);

  // 6) masked softmax over (src, type); per-type sums for agg_e
  k_softmax<<<Mc / 8, 256, 0, stream>>>(em, a_src, ae, w_bf16, wsum_e, hn,
                                        b_has_edge);

  // 7) agg_node = w @ node   [32 x 512x512x256]
  k_agg_gemm<<<dim3(Nc / 16, Hc / 16, Bc), 32, 0, stream>>>(w_bf16, node_swz,
                                                            agg_bf16);

  // 8) GAT projection + relu + select + scatter-add into out (atomic handles
  //    duplicate centers, matching .at[].add semantics)
  k_gat_gemm<<<dim3(Mc / 16, Hc / 16), 32, 0, stream>>>(
      agg_bf16, wg1_swz, wsum_e, ee_g, b_gat, hn, b_has_edge, node_f32,
      asp_st, opi_st, out);
}
